// GraphTransformer_60129542144693
// MI455X (gfx1250) — matrix operand order repaired and checked by reading, not compiled, not run on hardware
//
#include <hip/hip_runtime.h>
#include <hip/hip_bf16.h>
#include <math.h>

// ---- problem constants (match reference) ----
#define NNODES  20000
#define NEDGES  320000
#define DIMD    256
#define INNERD  128      // HEADS*DH
#define NHEADS  4
#define DHEAD   32
#define FFD     1024
#define NDEPTH  4

typedef __attribute__((ext_vector_type(16))) _Float16 v16h;
typedef __attribute__((ext_vector_type(8)))  _Float16 v8h;
typedef __attribute__((ext_vector_type(8)))  float    v8f;

// ---------------- small utility kernels ----------------

__global__ void f32_to_f16_kernel(const float* __restrict__ src,
                                  _Float16* __restrict__ dst, int n) {
    int i = blockIdx.x * blockDim.x + threadIdx.x;
    if (i < n) dst[i] = (_Float16)src[i];
}

__global__ void zero_f32_kernel(float* __restrict__ p, int n) {
    int i = blockIdx.x * blockDim.x + threadIdx.x;
    if (i < n) p[i] = 0.0f;
}

// ---------------- LayerNorm (fp32 in, f16 out) ----------------
__global__ void ln_kernel(const float* __restrict__ x,
                          const float* __restrict__ g,
                          const float* __restrict__ b,
                          _Float16* __restrict__ out) {
    const int row = blockIdx.x;
    const int t   = threadIdx.x;
    float v = x[(size_t)row * DIMD + t];
    __shared__ float s1[DIMD];
    __shared__ float s2[DIMD];
    s1[t] = v; s2[t] = v * v;
    __syncthreads();
    for (int off = DIMD / 2; off > 0; off >>= 1) {
        if (t < off) { s1[t] += s1[t + off]; s2[t] += s2[t + off]; }
        __syncthreads();
    }
    float mu  = s1[0] * (1.0f / DIMD);
    float var = s2[0] * (1.0f / DIMD) - mu * mu;
    float r   = rsqrtf(var + 1e-5f);
    out[(size_t)row * DIMD + t] = (_Float16)((v - mu) * r * g[t] + b[t]);
}

// ---------------- WMMA GEMM ----------------
// C[M x Nout] = A(M x K, f16 rm) * W(Nout x K, f16 rm)^T + bias
// block = 256 threads (8 waves). Block tile 32(M) x 128(N); wave tile 32x16:
// two 16x16x32 WMMAs per K-step sharing one B fragment. A tile staged into
// LDS with global_load_async_to_lds_b128 (double buffered, s_wait_asynccnt).
// grid.x = M/32, grid.y = Nout/128
// EPI: 0 = bias->f32 ; 1 = bias+GELU->f16 ; 2 = bias+residual->f32
template<int K, int EPI>
__global__ void gemm_wmma_kernel(const _Float16* __restrict__ A,
                                 const _Float16* __restrict__ W,
                                 const float* __restrict__ bias,
                                 const float* __restrict__ resid,
                                 float* __restrict__ out32,
                                 _Float16* __restrict__ out16,
                                 int ldn) {
    const int lane = threadIdx.x & 31;
    const int wave = threadIdx.x >> 5;
    const int m0 = blockIdx.x * 32;
    const int n0 = blockIdx.y * 128 + wave * 16;

    __shared__ __align__(16) _Float16 atile[2][32 * 32];  // 2 x 2KB
    const unsigned lds0 = (unsigned)(uintptr_t)(&atile[0][0]);

    const bool loader = (threadIdx.x < 128);   // waves 0..3 (wave-uniform)
    const int lr = threadIdx.x >> 2;           // row 0..31
    const int lc = (threadIdx.x & 3) << 3;     // half offset 0,8,16,24
    const _Float16* gsrc = A + (size_t)(m0 + lr) * K + lc;
    const unsigned ldsrow = (unsigned)(lr * 64 + (lc << 1));  // byte offset in tile

    const _Float16* brow = W + (size_t)(n0 + (lane & 15)) * K + ((lane >> 4) << 3);

    if (loader) {
        unsigned dst = lds0 + ldsrow;          // buffer 0
        asm volatile("global_load_async_to_lds_b128 %0, %1, off"
                     :: "v"(dst), "v"(gsrc) : "memory");
    }

    v8f acc[2] = {{}, {}};
    const int arow = (lane & 15);
    const int acol = ((lane >> 4) << 3);       // half offset 0 or 8

    for (int kk = 0; kk < K; kk += 32) {
        const int cur = (kk >> 5) & 1;
        if (kk + 32 < K) {
            if (loader) {
                unsigned dst = lds0 + (unsigned)((1 - cur) * 2048) + ldsrow;
                asm volatile("global_load_async_to_lds_b128 %0, %1, off"
                             :: "v"(dst), "v"(gsrc + kk + 32) : "memory");
                asm volatile("s_wait_asynccnt 0x1" ::: "memory");
            }
        } else {
            if (loader) asm volatile("s_wait_asynccnt 0x0" ::: "memory");
        }
        __syncthreads();

        const _Float16* ab = &atile[cur][0];
        __builtin_prefetch(brow + kk + 32, 0, 1);

        v8h blo = *(const v8h*)(brow + kk);
        v8h bhi = *(const v8h*)(brow + kk + 16);
        v16h bmat = __builtin_shufflevector(blo, bhi,
                        0,1,2,3,4,5,6,7,8,9,10,11,12,13,14,15);

        v8h a0lo = *(const v8h*)(ab + arow * 32 + acol);
        v8h a0hi = *(const v8h*)(ab + arow * 32 + acol + 16);
        v16h a0 = __builtin_shufflevector(a0lo, a0hi,
                        0,1,2,3,4,5,6,7,8,9,10,11,12,13,14,15);
        acc[0] = __builtin_amdgcn_wmma_f32_16x16x32_f16(false, a0, false, bmat,
                                                        (short)0, acc[0], false, false);

        v8h a1lo = *(const v8h*)(ab + (16 + arow) * 32 + acol);
        v8h a1hi = *(const v8h*)(ab + (16 + arow) * 32 + acol + 16);
        v16h a1 = __builtin_shufflevector(a1lo, a1hi,
                        0,1,2,3,4,5,6,7,8,9,10,11,12,13,14,15);
        acc[1] = __builtin_amdgcn_wmma_f32_16x16x32_f16(false, a1, false, bmat,
                                                        (short)0, acc[1], false, false);

        __syncthreads();
    }

    const int col  = n0 + (lane & 15);
    const float bcol = bias[col];
#pragma unroll
    for (int s = 0; s < 2; ++s) {
        const int rbase = m0 + s * 16 + ((lane >> 4) << 3);
#pragma unroll
        for (int r = 0; r < 8; ++r) {
            const size_t idx = (size_t)(rbase + r) * ldn + col;
            float val = acc[s][r] + bcol;
            if constexpr (EPI == 1) {
                val = 0.5f * val * (1.0f + erff(val * 0.7071067811865475f));
                out16[idx] = (_Float16)val;
            } else if constexpr (EPI == 2) {
                val += resid[idx];
                out32[idx] = val;
            } else {
                out32[idx] = val;
            }
        }
    }
}

// ---------------- edge attention kernels ----------------

__device__ __forceinline__ unsigned f2ord(float f) {
    unsigned u = __float_as_uint(f);
    return u ^ ((u >> 31) ? 0xFFFFFFFFu : 0x80000000u);
}
__device__ __forceinline__ float ord2f(unsigned u) {
    unsigned v = u ^ ((u >> 31) ? 0x80000000u : 0xFFFFFFFFu);
    return __uint_as_float(v);
}

__global__ void alpha_kernel(const int* __restrict__ ei,
                             const float* __restrict__ qkv,
                             float* __restrict__ alpha) {
    int tid = blockIdx.x * blockDim.x + threadIdx.x;
    if (tid >= NEDGES * NHEADS) return;
    int e = tid >> 2, h = tid & 3;
    int j = ei[e];
    int i = ei[NEDGES + e];
    const float4* q = (const float4*)(qkv + (size_t)i * 384 + h * DHEAD);
    const float4* k = (const float4*)(qkv + (size_t)j * 384 + INNERD + h * DHEAD);
    float s = 0.0f;
#pragma unroll
    for (int t = 0; t < DHEAD / 4; ++t) {
        float4 a = q[t], b = k[t];
        s += a.x * b.x + a.y * b.y + a.z * b.z + a.w * b.w;
    }
    alpha[tid] = s * 0.17677669529663689f;
}

__global__ void amax_kernel(const int* __restrict__ ei,
                            const float* __restrict__ alpha,
                            unsigned* __restrict__ amax) {
    int tid = blockIdx.x * blockDim.x + threadIdx.x;
    if (tid >= NEDGES * NHEADS) return;
    int e = tid >> 2, h = tid & 3;
    int i = ei[NEDGES + e];
    atomicMax(&amax[i * NHEADS + h], f2ord(alpha[tid]));
}

__global__ void exp_kernel(const int* __restrict__ ei,
                           float* __restrict__ alpha,
                           const unsigned* __restrict__ amax,
                           float* __restrict__ denom) {
    int tid = blockIdx.x * blockDim.x + threadIdx.x;
    if (tid >= NEDGES * NHEADS) return;
    int e = tid >> 2, h = tid & 3;
    int i = ei[NEDGES + e];
    float m  = ord2f(amax[i * NHEADS + h]);
    float ex = expf(alpha[tid] - m);
    alpha[tid] = ex;
    atomicAdd(&denom[i * NHEADS + h], ex);
}

__global__ void agg_kernel(const int* __restrict__ ei,
                           const float* __restrict__ qkv,
                           const float* __restrict__ ex,
                           const float* __restrict__ denom,
                           float* __restrict__ agg) {
    int tid = blockIdx.x * blockDim.x + threadIdx.x;
    if (tid >= NEDGES * INNERD) return;
    int e = tid >> 7;
    int r = tid & 127;
    int h = r >> 5;
    int j = ei[e];
    int i = ei[NEDGES + e];
    float w = ex[e * NHEADS + h] / (denom[i * NHEADS + h] + 1e-16f);
    atomicAdd(&agg[(size_t)i * INNERD + r],
              qkv[(size_t)j * 384 + 2 * INNERD + r] * w);
}

// ---------------- host orchestration ----------------

extern "C" void kernel_launch(void* const* d_in, const int* in_sizes, int n_in,
                              void* d_out, int out_size, void* d_ws, size_t ws_size,
                              hipStream_t stream) {
    const float* x     = (const float*)d_in[0];
    const int*   ei    = (const int*)  d_in[1];
    const float* ln1_g = (const float*)d_in[2];
    const float* ln1_b = (const float*)d_in[3];
    const float* wq    = (const float*)d_in[4];
    const float* bq    = (const float*)d_in[5];
    const float* wk    = (const float*)d_in[6];
    const float* bk    = (const float*)d_in[7];
    const float* wv    = (const float*)d_in[8];
    const float* bv    = (const float*)d_in[9];
    const float* wo    = (const float*)d_in[10];
    const float* bo    = (const float*)d_in[11];
    const float* ln2_g = (const float*)d_in[12];
    const float* ln2_b = (const float*)d_in[13];
    const float* w1    = (const float*)d_in[14];
    const float* b1    = (const float*)d_in[15];
    const float* w2    = (const float*)d_in[16];
    const float* b2    = (const float*)d_in[17];
    float* out = (float*)d_out;

    char* base = (char*)d_ws;
    _Float16* WF16  = (_Float16*)(base);               //  5,242,880 B
    _Float16* H16   = (_Float16*)(base +  5242880);    // 10,240,000 B
    float*    QKV   = (float*)   (base + 15482880);    // 30,720,000 B
    float*    ALPHA = (float*)   (base + 46202880);    //  5,120,000 B
    unsigned* AMAX  = (unsigned*)(base + 51322880);    //    320,000 B
    float*    DENOM = (float*)   (base + 51642880);    //    320,000 B
    float*    AGG   = (float*)   (base + 51962880);    // 10,240,000 B
    _Float16* AGG16 = (_Float16*)(base + 62202880);    //  5,120,000 B
    _Float16* FF116 = (_Float16*)(base + 15482880);    // overlays dead QKV region
    float*    BQKV  = (float*)   (base + 67322880);    //      6,144 B

    const size_t LSTR = 655360;
    const size_t OFF_QKV = 0, OFF_WO = 98304, OFF_W1 = 131072, OFF_W2 = 393216;

    const int CT = 256;
    auto cvt = [&](const float* s, _Float16* d, int n) {
        f32_to_f16_kernel<<<(n + CT - 1) / CT, CT, 0, stream>>>(s, d, n);
    };

    hipMemcpyAsync(out, x, (size_t)NNODES * DIMD * sizeof(float),
                   hipMemcpyDeviceToDevice, stream);

    for (int l = 0; l < NDEPTH; ++l) {
        _Float16* wl = WF16 + (size_t)l * LSTR;
        cvt(wq + (size_t)l * 32768,  wl + OFF_QKV,          32768);
        cvt(wk + (size_t)l * 32768,  wl + OFF_QKV + 32768,  32768);
        cvt(wv + (size_t)l * 32768,  wl + OFF_QKV + 65536,  32768);
        cvt(wo + (size_t)l * 32768,  wl + OFF_WO,           32768);
        cvt(w1 + (size_t)l * 262144, wl + OFF_W1,          262144);
        cvt(w2 + (size_t)l * 262144, wl + OFF_W2,          262144);
        hipMemcpyAsync(BQKV + l * 384,       bq + l * 128, 128 * sizeof(float),
                       hipMemcpyDeviceToDevice, stream);
        hipMemcpyAsync(BQKV + l * 384 + 128, bk + l * 128, 128 * sizeof(float),
                       hipMemcpyDeviceToDevice, stream);
        hipMemcpyAsync(BQKV + l * 384 + 256, bv + l * 128, 128 * sizeof(float),
                       hipMemcpyDeviceToDevice, stream);
    }

    const int EH  = NEDGES * NHEADS;
    const int EF  = NEDGES * INNERD;
    const dim3 gQKV(NNODES / 32, 384 / 128);
    const dim3 gO  (NNODES / 32, DIMD / 128);
    const dim3 gF1 (NNODES / 32, FFD / 128);

    for (int l = 0; l < NDEPTH; ++l) {
        const _Float16* wl = WF16 + (size_t)l * LSTR;

        ln_kernel<<<NNODES, DIMD, 0, stream>>>(out, ln1_g + l * DIMD, ln1_b + l * DIMD, H16);

        gemm_wmma_kernel<DIMD, 0><<<gQKV, 256, 0, stream>>>(
            H16, wl + OFF_QKV, BQKV + l * 384, nullptr, QKV, nullptr, 384);

        zero_f32_kernel<<<(NNODES * 136 + CT - 1) / CT, CT, 0, stream>>>(
            (float*)AMAX, NNODES * 136);

        alpha_kernel<<<(EH + CT - 1) / CT, CT, 0, stream>>>(ei, QKV, ALPHA);
        amax_kernel <<<(EH + CT - 1) / CT, CT, 0, stream>>>(ei, ALPHA, AMAX);
        exp_kernel  <<<(EH + CT - 1) / CT, CT, 0, stream>>>(ei, ALPHA, AMAX, DENOM);
        agg_kernel  <<<(EF + CT - 1) / CT, CT, 0, stream>>>(ei, QKV, ALPHA, DENOM, AGG);

        f32_to_f16_kernel<<<(NNODES * INNERD + CT - 1) / CT, CT, 0, stream>>>(
            AGG, AGG16, NNODES * INNERD);

        gemm_wmma_kernel<INNERD, 2><<<gO, 256, 0, stream>>>(
            AGG16, wl + OFF_WO, bo + l * DIMD, out, out, nullptr, DIMD);

        ln_kernel<<<NNODES, DIMD, 0, stream>>>(out, ln2_g + l * DIMD, ln2_b + l * DIMD, H16);

        gemm_wmma_kernel<DIMD, 1><<<gF1, 256, 0, stream>>>(
            H16, wl + OFF_W1, b1 + l * FFD, nullptr, nullptr, FF116, FFD);

        gemm_wmma_kernel<FFD, 2><<<gO, 256, 0, stream>>>(
            FF116, wl + OFF_W2, b2 + l * DIMD, out, out, nullptr, DIMD);
    }
}
